// PanopticSegment_50276887167232
// MI455X (gfx1250) — compile-verified, matrix operand order checked by reading, not run on hardware
//
#include <hip/hip_runtime.h>
#include <math.h>

typedef __attribute__((ext_vector_type(16))) _Float16 v16h;
typedef __attribute__((ext_vector_type(8)))  float    v8f;
typedef __attribute__((ext_vector_type(2)))  float    v2f;
typedef __attribute__((ext_vector_type(4)))  unsigned int u32x4;
typedef __attribute__((ext_vector_type(4)))  int      i32x4;
typedef __attribute__((ext_vector_type(8)))  int      i32x8;

#define NB 8
#define NC 21
#define HH 128
#define WW 128
#define HWD (HH*WW)      // 16384 pixels per image
#define MC 2048          // pre-NMS candidates
#define TOPK 500
#define NPAD 512         // TOPK padded to a multiple of 16 for WMMA N-tiles
#define IOU_THR 0.5f
#define PROB_THR 0.5f
#define COVC 10.0f

// ---------------------------------------------------------------------------
// Stage 1: fused softmax(max/argmax) + box decode.  Pure bandwidth stage.
// ---------------------------------------------------------------------------
__global__ void ps_decode(const float* __restrict__ mask,
                          const float* __restrict__ reg,
                          float* __restrict__ scores,
                          float* __restrict__ boxesAll) {
  int p = blockIdx.x * blockDim.x + threadIdx.x;
  if (p >= NB * HWD) return;
  int n = p / HWD;
  int rem = p - n * HWD;
  int y = rem >> 7;
  int x = rem & (WW - 1);

  const float* mb = mask + ((size_t)n * NC) * HWD + rem;
  float m = mb[0]; int cls = 0;
  #pragma unroll
  for (int c = 1; c < NC; ++c) {
    float v = mb[(size_t)c * HWD];
    if (v > m) { m = v; cls = c; }
  }
  float s = 0.f;
  #pragma unroll
  for (int c = 0; c < NC; ++c) s += __expf(mb[(size_t)c * HWD] - m);
  float prob = 1.0f / s;                 // max of softmax = exp(m-m)/sum
  bool frontal = (cls != 0) && (prob > PROB_THR);
  scores[p] = frontal ? prob : -1.0f;

  const float* rb = reg + ((size_t)n * 4) * HWD + rem;
  float cy = (float)y + tanhf(rb[0]);
  float cx = (float)x + tanhf(rb[(size_t)HWD]);
  float sy = fmaxf(rb[(size_t)2 * HWD], 4.0f);
  float sx = fmaxf(rb[(size_t)3 * HWD], 4.0f);
  float y1 = fminf(fmaxf(cy - sy, 0.f), 128.f);
  float x1 = fminf(fmaxf(cx - sx, 0.f), 128.f);
  float y2 = fminf(fmaxf(cy + sy, 0.f), 128.f);
  float x2 = fminf(fmaxf(cx + sx, 0.f), 128.f);
  float* bp = boxesAll + (size_t)p * 4;
  bp[0] = y1; bp[1] = x1; bp[2] = y2; bp[3] = x2;
}

// ---------------------------------------------------------------------------
// Stage 2: per-image descending bitonic sort of 16384 packed (score,idx)
// keys entirely in LDS (64 KB of the 320 KB WGP pool).  Replaces top_k(2048).
// ---------------------------------------------------------------------------
__global__ __launch_bounds__(1024)
void ps_sort(const float* __restrict__ scores,
             const float* __restrict__ boxesAll,
             float* __restrict__ candBoxes,
             float* __restrict__ candScores) {
  __shared__ unsigned int s[HWD];
  int n = blockIdx.x;
  int tid = threadIdx.x;
  const float* sc = scores + (size_t)n * HWD;

  for (int e = tid; e < HWD; e += 1024) {
    unsigned int b = __float_as_uint(sc[e]);
    unsigned int u = (b & 0x80000000u) ? ~b : (b | 0x80000000u);
    s[e] = (u & 0xFFFFC000u) | (unsigned int)e;
  }
  __syncthreads();

  for (int k = 2; k <= HWD; k <<= 1) {
    for (int j = k >> 1; j > 0; j >>= 1) {
      for (int t = tid; t < (HWD / 2); t += 1024) {
        int low = t & (j - 1);
        int i = ((t - low) << 1) | low;
        int pr = i | j;
        unsigned int a = s[i], b2 = s[pr];
        bool desc = (i & k) == 0;
        if (desc ? (a < b2) : (a > b2)) { s[i] = b2; s[pr] = a; }
      }
      __syncthreads();
    }
  }

  for (int e = tid; e < MC; e += 1024) {
    int idx = (int)(s[e] & 0x3FFFu);
    candScores[(size_t)n * MC + e] = sc[idx];
    const float* bp = boxesAll + ((size_t)n * HWD + idx) * 4;
    float* cp = candBoxes + ((size_t)n * MC + e) * 4;
    cp[0] = bp[0]; cp[1] = bp[1]; cp[2] = bp[2]; cp[3] = bp[3];
  }
}

// ---------------------------------------------------------------------------
// Stage 3: greedy NMS, one workgroup per image, all state in LDS (48 KB).
// ---------------------------------------------------------------------------
__global__ __launch_bounds__(256)
void ps_nms(const float* __restrict__ candBoxes,
            const float* __restrict__ candScores,
            int* __restrict__ keepOut) {
  __shared__ float4 sbox[MC];
  __shared__ float  sarea[MC];
  __shared__ int    skeep[MC];
  int n = blockIdx.x;
  int tid = threadIdx.x;

  for (int e = tid; e < MC; e += 256) {
    const float* bp = candBoxes + ((size_t)n * MC + e) * 4;
    float4 b = make_float4(bp[0], bp[1], bp[2], bp[3]);
    sbox[e] = b;
    sarea[e] = (b.z - b.x) * (b.w - b.y);
    skeep[e] = candScores[(size_t)n * MC + e] > 0.0f ? 1 : 0;
  }

  for (int i = 0; i < MC; ++i) {
    __syncthreads();
    if (!skeep[i]) continue;              // uniform branch across block
    float4 bi = sbox[i];
    float ai = sarea[i];
    for (int t = i + 1 + tid; t < MC; t += 256) {
      if (!skeep[t]) continue;
      float4 bj = sbox[t];
      float lty = fmaxf(bi.x, bj.x);
      float ltx = fmaxf(bi.y, bj.y);
      float rby = fminf(bi.z, bj.z);
      float rbx = fminf(bi.w, bj.w);
      float ih = fmaxf(rby - lty, 0.f);
      float iw = fmaxf(rbx - ltx, 0.f);
      float inter = ih * iw;
      float iou = inter / (ai + sarea[t] - inter + 1e-9f);
      if (iou > IOU_THR) skeep[t] = 0;
    }
  }
  __syncthreads();
  for (int e = tid; e < MC; e += 256) keepOut[(size_t)n * MC + e] = skeep[e];
}

// ---------------------------------------------------------------------------
// Stage 4: stable compaction + outputs + f32 B-matrix coefficient blocks.
// Each box slot: 8 f32 (32 B): [cy, by, cx, bx, e0, 0, 0, 0].  The zero pad
// lets hi-lane B-fragment reads be unconditional (they land on stored zeros).
// ---------------------------------------------------------------------------
__global__ __launch_bounds__(512)
void ps_select(const float* __restrict__ candBoxes,
               const int* __restrict__ keepIn,
               float* __restrict__ outBoxes,
               float* __restrict__ outValid,
               float* __restrict__ coeffs) {
  __shared__ int skeep[MC];
  __shared__ int slot[MC];
  int n = blockIdx.x;
  int tid = threadIdx.x;
  for (int e = tid; e < MC; e += 512) skeep[e] = keepIn[(size_t)n * MC + e];
  __syncthreads();
  if (tid == 0) {
    int nk = 0;
    for (int e = 0; e < MC; ++e) if (skeep[e])  slot[e] = nk++;
    for (int e = 0; e < MC; ++e) if (!skeep[e]) slot[e] = nk++;
  }
  __syncthreads();

  for (int e = tid; e < MC; e += 512) {
    int s = slot[e];
    if (s >= TOPK) continue;
    const float* bp = candBoxes + ((size_t)n * MC + e) * 4;
    float y1 = bp[0], x1 = bp[1], y2 = bp[2], x2 = bp[3];
    int kept = skeep[e];
    float* ob = outBoxes + ((size_t)n * TOPK + s) * 4;
    ob[0] = y1; ob[1] = x1; ob[2] = y2; ob[3] = x2;
    outValid[(size_t)n * TOPK + s] = kept ? 1.0f : 0.0f;
    float4* cp = (float4*)(coeffs + ((size_t)n * NPAD + s) * 8);
    if (kept) {
      float muy = 0.5f * (y1 + y2) - 64.0f;
      float mux = 0.5f * (x1 + x2) - 64.0f;
      float sy = fmaxf(y2 - y1, 1e-3f);
      float sx = fmaxf(x2 - x1, 1e-3f);
      float cyv = (COVC * COVC) / (sy * sy);
      float cxv = (COVC * COVC) / (sx * sx);
      float e0 = cyv * muy * muy + cxv * mux * mux;
      cp[0] = make_float4(cyv, -2.0f * muy * cyv, cxv, -2.0f * mux * cxv);
      cp[1] = make_float4(e0, 0.f, 0.f, 0.f);
    } else {
      cp[0] = make_float4(0.f, 0.f, 0.f, 0.f);
      cp[1] = make_float4(1.0e8f, 0.f, 0.f, 0.f);   // exp(-5e7) == 0
    }
  }
  if (tid >= TOPK && tid < NPAD) {         // pad slots [500,512)
    float4* cp = (float4*)(coeffs + ((size_t)n * NPAD + tid) * 8);
    cp[0] = make_float4(0.f, 0.f, 0.f, 0.f);
    cp[1] = make_float4(1.0e8f, 0.f, 0.f, 0.f);
  }
}

// ---------------------------------------------------------------------------
// Stage 5: Gaussian heatmap as a rank-5 GEMM on the matrix pipe, in f32.
// The image's 16 KB coefficient block is staged into LDS once per block by
// the Tensor Data Mover (1-D descriptor, 4096 x 4B, tile == tensor), waited
// with s_wait_tensorcnt; B fragments are unconditional ds_load_b64 reads.
// Two chained V_WMMA_F32_16X16X4_F32 per 16x16 tile (K=8 covers the rank-5
// quadratic); lanes 0-15 carry y-terms (K=0,1 / 4,5), lanes 16-31 x-terms
// (K=2,3 / 6,7) per the 32-bit A/B VGPR layout.  heat = max_N exp(-Q/2),
// column-max via width-16 shfl_xor (C/D layout: VGPR r = row r / r+8).
// ---------------------------------------------------------------------------
__global__ __launch_bounds__(128)
void ps_heat(const float* __restrict__ coeffs, float* __restrict__ heat) {
  __shared__ __align__(16) float scoef[NPAD * 8];   // 16 KB staging buffer

  int lane = threadIdx.x & 31;
  int wv = blockIdx.x * (blockDim.x >> 5) + (threadIdx.x >> 5);
  int img = wv >> 10;        // 1024 M-tiles per image; 4 waves/block same img
  int mt  = wv & 1023;

  const float* gsrc = coeffs + (size_t)img * NPAD * 8;

#if __has_builtin(__builtin_amdgcn_tensor_load_to_lds)
  if ((threadIdx.x >> 5) == 0) {          // one wave issues the DMA
    unsigned long long ga = (unsigned long long)(size_t)gsrc;
    unsigned int ldsOff = (unsigned int)(size_t)&scoef[0];
    u32x4 g0;
    g0[0] = 1u;                                          // count=1 (user D#)
    g0[1] = ldsOff;                                      // lds_addr (bytes)
    g0[2] = (unsigned int)(ga & 0xFFFFFFFFull);          // global_addr[31:0]
    g0[3] = (unsigned int)((ga >> 32) & 0x01FFFFFFull)   // global_addr[56:32]
            | 0x80000000u;                               // type=2 ("image")
    i32x8 g1;
    g1[0] = (int)(2u << 16);         // data_size = 4 B
    g1[1] = (int)(4096u << 16);      // tensor_dim0 = 4096 elements
    g1[2] = (int)(1u << 16);         // tensor_dim1 = 1
    g1[3] = (int)(4096u << 16);      // tile_dim0 = 4096
    g1[4] = 1;                       // tile_dim1 = 1
    g1[5] = 4096;                    // tensor_dim0_stride = 4096
    g1[6] = 0;
    g1[7] = 0;
    i32x4 g2 = {0, 0, 0, 0};
    i32x4 g3 = {0, 0, 0, 0};
#if defined(__clang_major__) && (__clang_major__ >= 23)
    i32x8 gz = {0, 0, 0, 0, 0, 0, 0, 0};
    __builtin_amdgcn_tensor_load_to_lds(g0, g1, g2, g3, gz, 0);
#else
    __builtin_amdgcn_tensor_load_to_lds(g0, g1, g2, g3, 0);
#endif
    __builtin_amdgcn_s_wait_tensorcnt(0);
  }
#else
  for (int e = (int)threadIdx.x; e < NPAD * 2; e += (int)blockDim.x)
    ((float4*)scoef)[e] = ((const float4*)gsrc)[e];
#endif
  __syncthreads();

  int m = lane & 15;
  int pm = mt * 16 + m;
  float fy = (float)(pm >> 7) - 64.0f;
  float fx = (float)(pm & (WW - 1)) - 64.0f;
  bool hiHalf = lane >= 16;
  float t = hiHalf ? fx : fy;           // hi lanes carry the x-axis K slots

  float rmax[8];
  #pragma unroll
  for (int r = 0; r < 8; ++r) rmax[r] = 0.0f;

#if __has_builtin(__builtin_amdgcn_wmma_f32_16x16x4_f32)
  // f32 matrix pipe: A step1 = (t^2, t), A step2 = (1,0)|(0,0)
  v2f a1; a1[0] = t * t; a1[1] = t;
  v2f a2; a2[0] = hiHalf ? 0.0f : 1.0f; a2[1] = 0.0f;
  int lsel = (lane >> 4) << 1;          // hi lanes read the (cx,bx)/(0,0) pair

  for (int nt = 0; nt < NPAD / 16; ++nt) {
    const float* sp = scoef + (size_t)(nt * 16 + m) * 8 + lsel;
    v2f b1 = *(const v2f*)(sp);         // lo: (cy,by)   hi: (cx,bx)
    v2f b2 = *(const v2f*)(sp + 4);     // lo: (e0,0)    hi: (0,0)
    v8f c = {};
    v8f d0 = __builtin_amdgcn_wmma_f32_16x16x4_f32(
        false, a1, false, b1, (short)0, c,  false, false);
    v8f d  = __builtin_amdgcn_wmma_f32_16x16x4_f32(
        false, a2, false, b2, (short)0, d0, false, false);
    #pragma unroll
    for (int r = 0; r < 8; ++r)
      rmax[r] = fmaxf(rmax[r], __expf(-0.5f * d[r]));   // fmaxf is NaN-safe
  }
#else
  // f16 fallback (K=5 of 32), converting from the f32 coefficient block.
  v16h a = {};
  if (!hiHalf) {
    a[0] = (_Float16)(fy * fy); a[1] = (_Float16)fy;
    a[2] = (_Float16)(fx * fx); a[3] = (_Float16)fx;
    a[4] = (_Float16)1.0f;
  }
  for (int nt = 0; nt < NPAD / 16; ++nt) {
    v16h bf = {};
    if (!hiHalf) {
      const float* sp = scoef + (size_t)(nt * 16 + m) * 8;
      bf[0] = (_Float16)sp[0]; bf[1] = (_Float16)sp[1];
      bf[2] = (_Float16)sp[2]; bf[3] = (_Float16)sp[3];
      bf[4] = (_Float16)sp[4];
    }
    v8f c = {};
    v8f d = __builtin_amdgcn_wmma_f32_16x16x32_f16(
        false, a, false, bf, (short)0, c, false, false);
    #pragma unroll
    for (int r = 0; r < 8; ++r)
      rmax[r] = fmaxf(rmax[r], __expf(-0.5f * d[r]));
  }
#endif

  // column-max across each 16-lane group (width-16 xor shuffles)
  #pragma unroll
  for (int off = 8; off >= 1; off >>= 1) {
    #pragma unroll
    for (int r = 0; r < 8; ++r)
      rmax[r] = fmaxf(rmax[r], __shfl_xor(rmax[r], off, 16));
  }
  if ((lane & 15) == 0) {
    int rowBase = mt * 16 + ((lane >> 4) << 3);  // lane0 -> rows 0..7, lane16 -> 8..15
    #pragma unroll
    for (int r = 0; r < 8; ++r)
      heat[(size_t)img * HWD + rowBase + r] = rmax[r];
  }
}

// ---------------------------------------------------------------------------
extern "C" void kernel_launch(void* const* d_in, const int* in_sizes, int n_in,
                              void* d_out, int out_size, void* d_ws, size_t ws_size,
                              hipStream_t stream) {
  const float* mask = (const float*)d_in[0];
  const float* reg  = (const float*)d_in[1];

  float* out = (float*)d_out;
  float* heat    = out;                                   // (8,1,128,128)
  float* maskOut = out + (size_t)NB * HWD;                // (8,21,128,128)
  float* boxOut  = maskOut + (size_t)NB * NC * HWD;       // (8,500,4)
  float* valOut  = boxOut + (size_t)NB * TOPK * 4;        // (8,500)

  float* ws = (float*)d_ws;
  float* scores     = ws;                                     // 131072 f32
  float* boxesAll   = scores + (size_t)NB * HWD;              // 524288 f32
  float* candBoxes  = boxesAll + (size_t)NB * HWD * 4;        // 65536 f32
  float* candScores = candBoxes + (size_t)NB * MC * 4;        // 16384 f32
  int*   keep       = (int*)(candScores + (size_t)NB * MC);   // 16384 i32
  float* coeffs     = (float*)(keep + (size_t)NB * MC);       // 32768 f32 (16B-aligned)

  // mask passthrough (output #2)
  hipMemcpyAsync(maskOut, mask, (size_t)NB * NC * HWD * sizeof(float),
                 hipMemcpyDeviceToDevice, stream);

  ps_decode<<<(NB * HWD) / 256, 256, 0, stream>>>(mask, reg, scores, boxesAll);
  ps_sort<<<NB, 1024, 0, stream>>>(scores, boxesAll, candBoxes, candScores);
  ps_nms<<<NB, 256, 0, stream>>>(candBoxes, candScores, keep);
  ps_select<<<NB, 512, 0, stream>>>(candBoxes, keep, boxOut, valOut, coeffs);
  ps_heat<<<(NB * 1024) / 4, 128, 0, stream>>>(coeffs, heat);
}
